// Attention_89928025244012
// MI455X (gfx1250) — matrix operand order repaired and checked by reading, not compiled, not run on hardware
//
#include <hip/hip_runtime.h>
#include <hip/hip_bf16.h>
#include <cstdint>

// ---------------------------------------------------------------------------
// Fused dual-stream MHA for MI455X (gfx1250, wave32, WMMA f32_16x16x32_f16)
//   x    = proj(softmax(QK^T/sqrt(d)) V )  @ Wp^T  + bp
//   x_im = proj(softmax(QK^T/sqrt(d)) Vi)  @ Wpi^T + bpi   (shared attn)
// CDNA5 paths: v_wmma_f32_16x16x32_f16, async-to-LDS DMA double buffering
// (global_load_async_to_lds_b128 + s_wait_asynccnt), global_prefetch.
// All kernels: 2-stage software pipeline, one barrier per k-step.
// ---------------------------------------------------------------------------

typedef __attribute__((ext_vector_type(16))) _Float16 v16h;
typedef __attribute__((ext_vector_type(8)))  _Float16 v8h;
typedef __attribute__((ext_vector_type(4)))  _Float16 v4h;
typedef __attribute__((ext_vector_type(8)))  float    v8f;

#define DIM       768
#define NSEQ      1024
#define BATCH     8
#define NHEAD     12
#define HDIM      64
#define ATT_SCALE 0.125f            // 64^-0.5
#define BNROWS    (BATCH * NSEQ)    // 8192
#define KSTEPS    (DIM / 32)        // 24

static __device__ __forceinline__ v16h cat8(v8h lo, v8h hi) {
  return __builtin_shufflevector(lo, hi, 0,1,2,3,4,5,6,7,8,9,10,11,12,13,14,15);
}

// 16x32 f16 A-fragment (contiguous K order, matches B-frag reads):
//   lanes 0-15 : M = lane,    K = {0..15}
//   lanes 16-31: M = lane-16, K = {16..31}
static __device__ __forceinline__ v16h load_a_frag(const _Float16* row32, int lhi) {
  v8h lo = *(const v8h*)(row32 + lhi * 16);
  v8h hi = *(const v8h*)(row32 + lhi * 16 + 8);
  return cat8(lo, hi);
}

static __device__ __forceinline__ v8f wmma_f16(v16h a, v16h b, v8f c) {
  return __builtin_amdgcn_wmma_f32_16x16x32_f16(false, a, false, b, (short)0, c,
                                                false, false);
}

// Async copy 16B global -> LDS (ASYNCcnt tracked, CDNA5 ISA).
static __device__ __forceinline__ void async_b128(uint32_t lds_off, const void* gptr) {
  const unsigned long long g = (unsigned long long)(uintptr_t)gptr;
  asm volatile("global_load_async_to_lds_b128 %0, %1, off"
               :: "v"(lds_off), "v"(g) : "memory");
}
static __device__ __forceinline__ void wait_async0() {
  asm volatile("s_wait_asynccnt 0x0" ::: "memory");
}
static __device__ __forceinline__ uint32_t lds_addr(const void* p) {
  return (uint32_t)(uintptr_t)p;   // addr[31:0] = LDS offset (ISA 10.2 aperture map)
}

// ---------------------------------------------------------------------------
// Kernel 1: input projection  Y[b*N+n][c] = sum_d X[n][b][d] * W[c][d]  (f16 out)
// 64x64x32 tile, 128 threads = 4 waves. Double-buffered LDS; next slab is
// register-prefetched (f32 source needs cvt) while WMMA consumes current slab.
// ---------------------------------------------------------------------------
__global__ __launch_bounds__(128)
void proj_gemm(const float* __restrict__ X, const float* __restrict__ W,
               _Float16* __restrict__ Y) {
  __shared__ __align__(32) _Float16 Xs[2][64][32];
  __shared__ __align__(32) _Float16 Ws[2][64][32];

  const int t    = threadIdx.x;
  const int wave = t >> 5;
  const int lane = t & 31;
  const int lrow = lane & 15;
  const int lhi  = lane >> 4;
  const int mbase = blockIdx.x * 64;   // output row tile (b*N+n)
  const int cbase = blockIdx.y * 64;   // output col tile

  float4 xr[4], wr[4];

  auto issue_loads = [&](int dk) {
#pragma unroll
    for (int i = 0; i < 4; ++i) {
      const int idx = i * 128 + t;            // 0..511 float4 chunks
      const int r = idx >> 3, c4 = (idx & 7) * 4;
      const int gr = mbase + r;               // = b*N + n
      const int bb = gr >> 10, n = gr & (NSEQ - 1);
      const size_t xrow = (size_t)(n * BATCH + bb) * DIM;
      const size_t wrow = (size_t)(cbase + r) * DIM;
      xr[i] = *(const float4*)&X[xrow + dk + c4];
      wr[i] = *(const float4*)&W[wrow + dk + c4];
      if (i == 0) {                           // prefetch slab after next
        const int dkn = (dk + 32) % DIM;
        __builtin_prefetch(&X[xrow + dkn + c4], 0, 1);
        __builtin_prefetch(&W[wrow + dkn + c4], 0, 1);
      }
    }
  };
  auto commit = [&](int buf) {
#pragma unroll
    for (int i = 0; i < 4; ++i) {
      const int idx = i * 128 + t;
      const int r = idx >> 3, c4 = (idx & 7) * 4;
      v4h xh, wh;
      xh[0] = (_Float16)xr[i].x; xh[1] = (_Float16)xr[i].y;
      xh[2] = (_Float16)xr[i].z; xh[3] = (_Float16)xr[i].w;
      wh[0] = (_Float16)wr[i].x; wh[1] = (_Float16)wr[i].y;
      wh[2] = (_Float16)wr[i].z; wh[3] = (_Float16)wr[i].w;
      *(v4h*)&Xs[buf][r][c4] = xh;
      *(v4h*)&Ws[buf][r][c4] = wh;
    }
  };

  issue_loads(0);
  commit(0);

  v8f acc[4] = {v8f{}, v8f{}, v8f{}, v8f{}};
  for (int s = 0; s < KSTEPS; ++s) {
    const int buf = s & 1;
    __syncthreads();                           // buf's stores visible block-wide
    if (s + 1 < KSTEPS) issue_loads((s + 1) * 32);   // overlap with compute

    const v16h a = load_a_frag(&Xs[buf][16 * wave + lrow][0], lhi);
    v16h bf[4];
#pragma unroll
    for (int j = 0; j < 4; ++j)
      bf[j] = *(const v16h*)(&Ws[buf][j * 16 + lrow][lhi * 16]);
#pragma unroll
    for (int j = 0; j < 4; ++j)
      acc[j] = wmma_f16(a, bf[j], acc[j]);

    if (s + 1 < KSTEPS) commit(buf ^ 1);       // other buffer: no barrier needed
  }

#pragma unroll
  for (int j = 0; j < 4; ++j)
#pragma unroll
    for (int r = 0; r < 8; ++r) {
      const int grow = mbase + 16 * wave + r + 8 * lhi;  // C layout: VGPR r -> M=r(+8)
      const int gcol = cbase + j * 16 + lrow;
      Y[(size_t)grow * DIM + gcol] = (_Float16)acc[j][r];
    }
}

// ---------------------------------------------------------------------------
// Kernel 2: flash attention with shared softmax, two value streams.
// grid = (N/64, H, B); block = 128 = 4 waves; wave w owns query rows 16w..16w+15.
// Pipeline: next K tile DMA'd async into LDS[buf^1] and next V/Vi rows loaded
// into registers while current tile's S/softmax/PV WMMAs run.
// ---------------------------------------------------------------------------
__global__ __launch_bounds__(128)
void attn_flash(const _Float16* __restrict__ Q, const _Float16* __restrict__ K,
                const _Float16* __restrict__ V, const _Float16* __restrict__ Vi,
                _Float16* __restrict__ O, _Float16* __restrict__ Oi) {
  __shared__ __align__(32) _Float16 Ks  [2][32][64];    // K tile row-major [key][d]
  __shared__ __align__(32) _Float16 Vst [2][64][48];    // V  tile transposed [d][key]
  __shared__ __align__(32) _Float16 Vist[2][64][48];    // Vi tile transposed
  __shared__ __align__(32) _Float16 Ps  [4][16][32];    // per-wave P slab

  const int t    = threadIdx.x;
  const int wave = t >> 5;
  const int lane = t & 31;
  const int lrow = lane & 15;
  const int lhi  = lane >> 4;

  const int qbase = blockIdx.x * 64;
  const int h     = blockIdx.y;
  const int b     = blockIdx.z;

  const _Float16* qptr  = Q  + (size_t)b * NSEQ * DIM + h * HDIM;
  const _Float16* kptr  = K  + (size_t)b * NSEQ * DIM + h * HDIM;
  const _Float16* vptr  = V  + (size_t)b * NSEQ * DIM + h * HDIM;
  const _Float16* viptr = Vi + (size_t)b * NSEQ * DIM + h * HDIM;

  // Q A-fragments for this wave's 16 rows, both 32-wide d chunks -> registers.
  const int qrow = qbase + 16 * wave + lrow;
  const v16h aq0 = load_a_frag(qptr + (size_t)qrow * DIM + 0,  lhi);
  const v16h aq1 = load_a_frag(qptr + (size_t)qrow * DIM + 32, lhi);

  const uint32_t ks_lds = lds_addr(&Ks[0][0][0]);

  v8h vr[2], vir[2];                          // V/Vi prefetch registers

  auto issue_k_async = [&](int kt, int buf) { // byte-exact copy -> async DMA
#pragma unroll
    for (int i = 0; i < 2; ++i) {
      const int idx = i * 128 + t;            // 0..255 b128 chunks
      const int row = idx >> 3, c8 = (idx & 7) * 8;
      async_b128(ks_lds + (uint32_t)(buf * 4096 + (row * 64 + c8) * 2),
                 kptr + (size_t)(kt + row) * DIM + c8);
    }
  };
  auto issue_v_loads = [&](int kt) {
#pragma unroll
    for (int i = 0; i < 2; ++i) {
      const int idx = i * 128 + t;            // 0..255 v8h chunks
      const int kl = idx >> 3, d8 = (idx & 7) * 8;
      vr[i]  = *(const v8h*)(vptr  + (size_t)(kt + kl) * DIM + d8);
      vir[i] = *(const v8h*)(viptr + (size_t)(kt + kl) * DIM + d8);
    }
  };
  auto commit_v = [&](int buf) {              // transposed scatter for B-frags
#pragma unroll
    for (int i = 0; i < 2; ++i) {
      const int idx = i * 128 + t;
      const int kl = idx >> 3, d8 = (idx & 7) * 8;
#pragma unroll
      for (int e = 0; e < 8; ++e) {
        Vst [buf][d8 + e][kl] = vr[i][e];
        Vist[buf][d8 + e][kl] = vir[i][e];
      }
    }
  };

  float m_r[8], l_r[8];
#pragma unroll
  for (int r = 0; r < 8; ++r) { m_r[r] = -1e30f; l_r[r] = 0.f; }
  v8f o0[4] = {v8f{}, v8f{}, v8f{}, v8f{}};
  v8f o1[4] = {v8f{}, v8f{}, v8f{}, v8f{}};

  // Prologue: stage tile 0 into buffer 0.
  issue_k_async(0, 0);
  issue_v_loads(0);
  commit_v(0);
  wait_async0();

  for (int kt = 0; kt < NSEQ; kt += 32) {
    const int  buf  = (kt >> 5) & 1;
    const bool more = (kt + 32 < NSEQ);
    __syncthreads();                           // buf's K (async) + V (ds) visible
    if (more) { issue_k_async(kt + 32, buf ^ 1); issue_v_loads(kt + 32); }

    // ---- S = Q K^T for keys [kt, kt+32): two 16-col C tiles --------------
    // B-frag (32x16): col = lane&15 (key), k-rows = lhi*16.. (contiguous d)
    const v16h bk00 = *(const v16h*)(&Ks[buf][lrow][lhi * 16]);
    const v16h bk01 = *(const v16h*)(&Ks[buf][lrow][32 + lhi * 16]);
    const v16h bk10 = *(const v16h*)(&Ks[buf][16 + lrow][lhi * 16]);
    const v16h bk11 = *(const v16h*)(&Ks[buf][16 + lrow][32 + lhi * 16]);
    v8f s0 = v8f{}, s1 = v8f{};
    s0 = wmma_f16(aq0, bk00, s0);
    s0 = wmma_f16(aq1, bk01, s0);
    s1 = wmma_f16(aq0, bk10, s1);
    s1 = wmma_f16(aq1, bk11, s1);

    // ---- streaming softmax update ---------------------------------------
    float mt[8], p0[8], p1[8], alpha[8], rs[8];
#pragma unroll
    for (int r = 0; r < 8; ++r) mt[r] = fmaxf(s0[r], s1[r]) * ATT_SCALE;
#pragma unroll
    for (int msk = 1; msk < 16; msk <<= 1)
#pragma unroll
      for (int r = 0; r < 8; ++r) mt[r] = fmaxf(mt[r], __shfl_xor(mt[r], msk, 32));
#pragma unroll
    for (int r = 0; r < 8; ++r) {
      const float mn = fmaxf(m_r[r], mt[r]);
      alpha[r] = __expf(m_r[r] - mn);
      m_r[r]   = mn;
      p0[r] = __expf(s0[r] * ATT_SCALE - mn);
      p1[r] = __expf(s1[r] * ATT_SCALE - mn);
      rs[r] = p0[r] + p1[r];
    }
#pragma unroll
    for (int msk = 1; msk < 16; msk <<= 1)
#pragma unroll
      for (int r = 0; r < 8; ++r) rs[r] += __shfl_xor(rs[r], msk, 32);
#pragma unroll
    for (int r = 0; r < 8; ++r) l_r[r] = l_r[r] * alpha[r] + rs[r];
#pragma unroll
    for (int j = 0; j < 4; ++j)
#pragma unroll
      for (int r = 0; r < 8; ++r) { o0[j][r] *= alpha[r]; o1[j][r] *= alpha[r]; }

    // ---- C-layout -> A-layout reshape of P via per-wave LDS slab ---------
#pragma unroll
    for (int r = 0; r < 8; ++r) {
      Ps[wave][r + 8 * lhi][lrow]      = (_Float16)p0[r];
      Ps[wave][r + 8 * lhi][16 + lrow] = (_Float16)p1[r];
    }
    const v16h ap = load_a_frag(&Ps[wave][lrow][0], lhi);  // wave-private slab

    // ---- O += P * V (both streams share P) -------------------------------
#pragma unroll
    for (int j = 0; j < 4; ++j) {
      const v16h bv  = *(const v16h*)(&Vst [buf][j * 16 + lrow][lhi * 16]);
      const v16h bvi = *(const v16h*)(&Vist[buf][j * 16 + lrow][lhi * 16]);
      o0[j] = wmma_f16(ap, bv,  o0[j]);
      o1[j] = wmma_f16(ap, bvi, o1[j]);
    }

    if (more) commit_v(buf ^ 1);               // other buffer: no barrier needed
    wait_async0();                             // this wave's DMA into buf^1 done
  }

  // ---- normalize and store (f16, merged-head layout) ------------------------
  float inv[8];
#pragma unroll
  for (int r = 0; r < 8; ++r) inv[r] = 1.0f / l_r[r];
#pragma unroll
  for (int j = 0; j < 4; ++j)
#pragma unroll
    for (int r = 0; r < 8; ++r) {
      const size_t orow = (size_t)(b * NSEQ + qbase + 16 * wave + r + 8 * lhi);
      const int    ocol = h * HDIM + j * 16 + lrow;
      O [orow * DIM + ocol] = (_Float16)(o0[j][r] * inv[r]);
      Oi[orow * DIM + ocol] = (_Float16)(o1[j][r] * inv[r]);
    }
}

// ---------------------------------------------------------------------------
// Kernel 3: output projection  out[r][c] = sum_d Yin[r][d]*W[c][d] + bias[c]
// X tile (f16): async DMA straight into the ping-pong buffer (the DMA *is*
// the prefetch). W tile (f32): register prefetch + cvt.
// ---------------------------------------------------------------------------
__global__ __launch_bounds__(128)
void outproj_gemm(const _Float16* __restrict__ Yin, const float* __restrict__ W,
                  const float* __restrict__ bias, float* __restrict__ out) {
  __shared__ __align__(32) _Float16 Xs[2][64][32];
  __shared__ __align__(32) _Float16 Ws[2][64][32];

  const int t    = threadIdx.x;
  const int wave = t >> 5;
  const int lane = t & 31;
  const int lrow = lane & 15;
  const int lhi  = lane >> 4;
  const int mbase = blockIdx.x * 64;
  const int cbase = blockIdx.y * 64;
  const uint32_t xs_lds = lds_addr(&Xs[0][0][0]);

  float4 wr[4];

  auto issue_x_async = [&](int dk, int buf) {
#pragma unroll
    for (int i = 0; i < 2; ++i) {
      const int idx = i * 128 + t;            // 0..255 b128 chunks
      const int r = idx >> 2, c8 = (idx & 3) * 8;
      async_b128(xs_lds + (uint32_t)(buf * 4096 + (r * 32 + c8) * 2),
                 Yin + (size_t)(mbase + r) * DIM + dk + c8);
    }
  };
  auto issue_w_loads = [&](int dk) {
#pragma unroll
    for (int i = 0; i < 4; ++i) {
      const int idx = i * 128 + t;
      const int r = idx >> 3, c4 = (idx & 7) * 4;
      wr[i] = *(const float4*)&W[(size_t)(cbase + r) * DIM + dk + c4];
    }
  };
  auto commit_w = [&](int buf) {
#pragma unroll
    for (int i = 0; i < 4; ++i) {
      const int idx = i * 128 + t;
      const int r = idx >> 3, c4 = (idx & 7) * 4;
      v4h wh;
      wh[0] = (_Float16)wr[i].x; wh[1] = (_Float16)wr[i].y;
      wh[2] = (_Float16)wr[i].z; wh[3] = (_Float16)wr[i].w;
      *(v4h*)&Ws[buf][r][c4] = wh;
    }
  };

  issue_x_async(0, 0);
  issue_w_loads(0);
  commit_w(0);
  wait_async0();

  v8f acc[4] = {v8f{}, v8f{}, v8f{}, v8f{}};
  for (int s = 0; s < KSTEPS; ++s) {
    const int buf = s & 1;
    __syncthreads();
    if (s + 1 < KSTEPS) {
      issue_x_async((s + 1) * 32, buf ^ 1);    // DMA next slab during compute
      issue_w_loads((s + 1) * 32);
    }

    const v16h a = load_a_frag(&Xs[buf][16 * wave + lrow][0], lhi);
    v16h bf[4];
#pragma unroll
    for (int j = 0; j < 4; ++j)
      bf[j] = *(const v16h*)(&Ws[buf][j * 16 + lrow][lhi * 16]);
#pragma unroll
    for (int j = 0; j < 4; ++j)
      acc[j] = wmma_f16(a, bf[j], acc[j]);

    if (s + 1 < KSTEPS) commit_w(buf ^ 1);
    wait_async0();
  }

#pragma unroll
  for (int j = 0; j < 4; ++j)
#pragma unroll
    for (int r = 0; r < 8; ++r) {
      const int grow = mbase + 16 * wave + r + 8 * lhi;
      const int gcol = cbase + j * 16 + lrow;
      out[(size_t)grow * DIM + gcol] = acc[j][r] + bias[gcol];
    }
}

// ---------------------------------------------------------------------------
// Host launcher
// ---------------------------------------------------------------------------
extern "C" void kernel_launch(void* const* d_in, const int* in_sizes, int n_in,
                              void* d_out, int out_size, void* d_ws, size_t ws_size,
                              hipStream_t stream) {
  const float* q    = (const float*)d_in[0];
  const float* k    = (const float*)d_in[1];
  const float* v    = (const float*)d_in[2];
  const float* vimg = (const float*)d_in[3];
  const float* Wq   = (const float*)d_in[4];
  const float* Wk   = (const float*)d_in[5];
  const float* Wv   = (const float*)d_in[6];
  const float* Wvim = (const float*)d_in[7];
  const float* Wp   = (const float*)d_in[8];
  const float* bp   = (const float*)d_in[9];
  const float* Wpi  = (const float*)d_in[10];
  const float* bpi  = (const float*)d_in[11];
  float* out = (float*)d_out;

  _Float16* ws = (_Float16*)d_ws;
  const size_t SB = (size_t)BNROWS * DIM;  // 6,291,456 halves per buffer
  _Float16* pQ  = ws;
  _Float16* pK  = ws + 1 * SB;
  _Float16* pV  = ws + 2 * SB;
  _Float16* pVi = ws + 3 * SB;
  _Float16* pO  = ws + 4 * SB;
  _Float16* pOi = ws + 5 * SB;

  const dim3 blk(128);
  const dim3 gemm_grid(BNROWS / 64, DIM / 64);       // (128, 12)
  const dim3 attn_grid(NSEQ / 64, NHEAD, BATCH);     // (16, 12, 8)

  proj_gemm<<<gemm_grid, blk, 0, stream>>>(q,    Wq,   pQ);
  proj_gemm<<<gemm_grid, blk, 0, stream>>>(k,    Wk,   pK);
  proj_gemm<<<gemm_grid, blk, 0, stream>>>(v,    Wv,   pV);
  proj_gemm<<<gemm_grid, blk, 0, stream>>>(vimg, Wvim, pVi);

  attn_flash<<<attn_grid, blk, 0, stream>>>(pQ, pK, pV, pVi, pO, pOi);

  outproj_gemm<<<gemm_grid, blk, 0, stream>>>(pO,  Wp,  bp,  out);
  outproj_gemm<<<gemm_grid, blk, 0, stream>>>(pOi, Wpi, bpi, out + SB);
}
